// ChebyNet_27462020891065
// MI455X (gfx1250) — matrix operand order repaired and checked by reading, not compiled, not run on hardware
//
#include <hip/hip_runtime.h>

// ---------------- types ----------------
typedef float  v8f   __attribute__((ext_vector_type(8)));
typedef __bf16 v16bf __attribute__((ext_vector_type(16)));
typedef __bf16 v8bf  __attribute__((ext_vector_type(8)));

#define NNODES 50000
#define NEDGES 400000
#define D_IN   128
#define H1     512
#define H2     300
#define BN_EPS 1e-5f
// lambda_max = 2.0 -> scale = 1.0, loop (self) coefficient = 0.0
#define CHEB_SCALE 1.0f
#define CHEB_LOOP  0.0f

__device__ __forceinline__ unsigned short f32_to_bf16_rne(float f) {
  unsigned int u = __float_as_uint(f);
  unsigned int r = u + 0x7FFFu + ((u >> 16) & 1u);
  return (unsigned short)(r >> 16);
}

// ---------------- small utility kernels ----------------
__global__ void zero_f32(float* __restrict__ p, long long n) {
  long long i = (long long)blockIdx.x * blockDim.x + threadIdx.x;
  long long stride = (long long)gridDim.x * blockDim.x;
  for (; i < n; i += stride) p[i] = 0.0f;
}

__global__ void deg_kernel(const long long* __restrict__ rows,
                           const float* __restrict__ ew,
                           float* __restrict__ deg, int E) {
  int e = blockIdx.x * blockDim.x + threadIdx.x;
  if (e < E) atomicAdd(&deg[(int)rows[e]], ew[e]);
}

__global__ void dinv_kernel(const float* __restrict__ deg,
                            float* __restrict__ dinv, int n) {
  int i = blockIdx.x * blockDim.x + threadIdx.x;
  if (i < n) {
    float d = deg[i];
    dinv[i] = (d > 0.0f) ? rsqrtf(d) : 0.0f;
  }
}

__global__ void wedge_kernel(const long long* __restrict__ rows,
                             const long long* __restrict__ cols,
                             const float* __restrict__ ew,
                             const float* __restrict__ dinv,
                             float* __restrict__ w, int E) {
  int e = blockIdx.x * blockDim.x + threadIdx.x;
  if (e < E)
    w[e] = -dinv[(int)rows[e]] * ew[e] * dinv[(int)cols[e]] * CHEB_SCALE;
}

__global__ void cvt_bf16_kernel(const float* __restrict__ in,
                                unsigned short* __restrict__ out, long long n) {
  long long i = (long long)blockIdx.x * blockDim.x + threadIdx.x;
  long long stride = (long long)gridDim.x * blockDim.x;
  for (; i < n; i += stride) out[i] = f32_to_bf16_rne(in[i]);
}

// tx1 = agg + loop * x  (bf16 output for the GEMM A-operand)
__global__ void finalize_tx_kernel(const float* __restrict__ agg,
                                   const float* __restrict__ x,
                                   unsigned short* __restrict__ out,
                                   long long n, float loopc) {
  long long i = (long long)blockIdx.x * blockDim.x + threadIdx.x;
  long long stride = (long long)gridDim.x * blockDim.x;
  for (; i < n; i += stride) out[i] = f32_to_bf16_rne(agg[i] + loopc * x[i]);
}

// W[K][Ncols] (f32, row-major) -> Wt[Ncols][K] (bf16)
__global__ void transpose_bf16_kernel(const float* __restrict__ W,
                                      unsigned short* __restrict__ Wt,
                                      int K, int Ncols) {
  long long total = (long long)K * Ncols;
  long long i = (long long)blockIdx.x * blockDim.x + threadIdx.x;
  long long stride = (long long)gridDim.x * blockDim.x;
  for (; i < total; i += stride) {
    int n = (int)(i / K);
    int k = (int)(i - (long long)n * K);
    Wt[i] = f32_to_bf16_rne(W[(long long)k * Ncols + n]);
  }
}

// ---------------- SpMM: out[dst] += w[e] * X[src]  (1 wave per edge) ----------------
__global__ __launch_bounds__(256) void spmm_edge_atomic(
    const long long* __restrict__ rows, const long long* __restrict__ cols,
    const float* __restrict__ w, const float* __restrict__ X,
    float* __restrict__ out, int E, int D) {
  int wave = (blockIdx.x * blockDim.x + threadIdx.x) >> 5;
  if (wave >= E) return;                 // wave-uniform exit
  int lane = threadIdx.x & 31;
  int src = (int)rows[wave];
  int dst = (int)cols[wave];
  float ww = w[wave];
  const float* xp = X + (long long)src * D;
  float* op = out + (long long)dst * D;
  for (int d = lane * 4; d < D; d += 128) {
    float4 v = *reinterpret_cast<const float4*>(xp + d);
    atomicAdd(op + d + 0, ww * v.x);
    atomicAdd(op + d + 1, ww * v.y);
    atomicAdd(op + d + 2, ww * v.z);
    atomicAdd(op + d + 3, ww * v.w);
  }
}

// ---------------- WMMA fragment helpers ----------------
__device__ __forceinline__ v16bf load_frag16(const unsigned short* p,
                                             const unsigned short* q) {
  v8bf lo = *reinterpret_cast<const v8bf*>(p);
  v8bf hi = *reinterpret_cast<const v8bf*>(q);
  v16bf r;
#pragma unroll
  for (int i = 0; i < 8; ++i) { r[i] = lo[i]; r[i + 8] = hi[i]; }
  return r;
}

// A 16x32 bf16 fragment at k0: lane (h,l): elems 0..7 -> K=k0+8h+e, 8..15 -> K=k0+16+8h+e
__device__ __forceinline__ v16bf load_fragA(const unsigned short* aBase, int k0, int h) {
  return load_frag16(aBase + k0 + h * 8, aBase + k0 + 16 + h * 8);
}
// B 32x16 bf16 fragment at k0: lane (h,l): elems 0..7 -> K=k0+8h+e, 8..15 -> K=k0+16+8h+e, col=l
__device__ __forceinline__ v16bf load_fragB(const unsigned short* bBase, int k0, int h) {
  return load_frag16(bBase + k0 + h * 8, bBase + k0 + 16 + h * 8);
}

#define WMMA_BF16(A_, B_, C_) \
  __builtin_amdgcn_wmma_f32_16x16x32_bf16(false, (A_), false, (B_), (short)0, (C_), false, false)

// ---------------- dual-K WMMA GEMM: C = A0@B0 + A1@B1 + bias ----------------
// A: [M][K] bf16 row-major.  Bt: [>=N][K] bf16 (pre-transposed weights).
// One wave computes a 32(M) x 64(N) tile: 2 A-fragments x 4 B-fragments, 8 f32
// accumulators. K loop manually unrolled 2x with role-alternating buffers
// (X consumed while Y loads, then Y consumed while X reloads) so the double
// buffer never needs register-to-register rotation. Requires K % 64 == 0.
__global__ __launch_bounds__(256) void gemm_dualK_wmma(
    const unsigned short* __restrict__ A0, const unsigned short* __restrict__ B0t,
    const unsigned short* __restrict__ A1, const unsigned short* __restrict__ B1t,
    const float* __restrict__ bias, float* __restrict__ C,
    int M, int N, int K) {
  const int nTiles = (N + 63) >> 6;
  const int mTiles = (M + 31) >> 5;      // 32-row tiles, last may be partial
  const int wave = (blockIdx.x * blockDim.x + threadIdx.x) >> 5;
  if (wave >= nTiles * mTiles) return;   // wave-uniform exit
  const int lane = threadIdx.x & 31;
  const int m0 = (wave / nTiles) << 5;
  const int n0 = (wave % nTiles) << 6;
  const int h = lane >> 4;               // half-wave select
  const int l = lane & 15;

  // clamp A rows for the M tail (stores masked later; EXEC stays all-1)
  int arow0 = m0 + l;      if (arow0 > M - 1) arow0 = M - 1;
  int arow1 = m0 + 16 + l; if (arow1 > M - 1) arow1 = M - 1;
  // clamp B columns for the N tail
  int ncol[4];
#pragma unroll
  for (int s = 0; s < 4; ++s) {
    int n = n0 + s * 16 + l;
    ncol[s] = (n < N) ? n : (N - 1);
  }

  v8f acc[8];
#pragma unroll
  for (int s = 0; s < 8; ++s) acc[s] = (v8f){0.f,0.f,0.f,0.f,0.f,0.f,0.f,0.f};

  for (int pass = 0; pass < 2; ++pass) {
    const unsigned short* A  = pass ? A1 : A0;
    const unsigned short* Bt = pass ? B1t : B0t;
    const unsigned short* aB0 = A + (long long)arow0 * K;
    const unsigned short* aB1 = A + (long long)arow1 * K;
    const unsigned short* bB0 = Bt + (long long)ncol[0] * K;
    const unsigned short* bB1 = Bt + (long long)ncol[1] * K;
    const unsigned short* bB2 = Bt + (long long)ncol[2] * K;
    const unsigned short* bB3 = Bt + (long long)ncol[3] * K;

    // buffer X <- K-step 0
    v16bf xa0 = load_fragA(aB0, 0, h);
    v16bf xa1 = load_fragA(aB1, 0, h);
    v16bf xb0 = load_fragB(bB0, 0, h);
    v16bf xb1 = load_fragB(bB1, 0, h);
    v16bf xb2 = load_fragB(bB2, 0, h);
    v16bf xb3 = load_fragB(bB3, 0, h);

    for (int k0 = 0; k0 < K; k0 += 64) {
      // buffer Y <- K-step k0+32 (always in range: K % 64 == 0)
      const int ky = k0 + 32;
      v16bf ya0 = load_fragA(aB0, ky, h);
      v16bf ya1 = load_fragA(aB1, ky, h);
      v16bf yb0 = load_fragB(bB0, ky, h);
      v16bf yb1 = load_fragB(bB1, ky, h);
      v16bf yb2 = load_fragB(bB2, ky, h);
      v16bf yb3 = load_fragB(bB3, ky, h);

      // consume X (k0)
      acc[0] = WMMA_BF16(xa0, xb0, acc[0]);
      acc[1] = WMMA_BF16(xa0, xb1, acc[1]);
      acc[2] = WMMA_BF16(xa0, xb2, acc[2]);
      acc[3] = WMMA_BF16(xa0, xb3, acc[3]);
      acc[4] = WMMA_BF16(xa1, xb0, acc[4]);
      acc[5] = WMMA_BF16(xa1, xb1, acc[5]);
      acc[6] = WMMA_BF16(xa1, xb2, acc[6]);
      acc[7] = WMMA_BF16(xa1, xb3, acc[7]);

      // buffer X <- K-step k0+64 (uniform guard; skipped on last iteration)
      const int kx = k0 + 64;
      if (kx < K) {
        xa0 = load_fragA(aB0, kx, h);
        xa1 = load_fragA(aB1, kx, h);
        xb0 = load_fragB(bB0, kx, h);
        xb1 = load_fragB(bB1, kx, h);
        xb2 = load_fragB(bB2, kx, h);
        xb3 = load_fragB(bB3, kx, h);
      }

      // consume Y (k0+32)
      acc[0] = WMMA_BF16(ya0, yb0, acc[0]);
      acc[1] = WMMA_BF16(ya0, yb1, acc[1]);
      acc[2] = WMMA_BF16(ya0, yb2, acc[2]);
      acc[3] = WMMA_BF16(ya0, yb3, acc[3]);
      acc[4] = WMMA_BF16(ya1, yb0, acc[4]);
      acc[5] = WMMA_BF16(ya1, yb1, acc[5]);
      acc[6] = WMMA_BF16(ya1, yb2, acc[6]);
      acc[7] = WMMA_BF16(ya1, yb3, acc[7]);
    }
  }

  // bias + store (C/D layout: VGPR j -> row 8h + j, col = 16s + l within tile)
#pragma unroll
  for (int s = 0; s < 4; ++s) {
    int n = n0 + s * 16 + l;
    if (n < N) {
      float bv = bias[n];
#pragma unroll
      for (int half = 0; half < 2; ++half) {
#pragma unroll
        for (int j = 0; j < 8; ++j) {
          int row = m0 + half * 16 + h * 8 + j;
          if (row < M)
            C[(long long)row * N + n] = acc[half * 4 + s][j] + bv;
        }
      }
    }
  }
}

// ---------------- BatchNorm ----------------
__global__ void bn_stats_kernel(const float* __restrict__ H,
                                float* __restrict__ sum, float* __restrict__ sumsq,
                                int M, int Ncols, int rowsPerBlock) {
  int j = blockIdx.x * blockDim.x + threadIdx.x;
  if (j >= Ncols) return;
  int r0 = blockIdx.y * rowsPerBlock;
  int r1 = r0 + rowsPerBlock; if (r1 > M) r1 = M;
  float s = 0.f, s2 = 0.f;
  for (int r = r0; r < r1; ++r) {
    float v = H[(long long)r * Ncols + j];
    s += v; s2 += v * v;
  }
  atomicAdd(&sum[j], s);
  atomicAdd(&sumsq[j], s2);
}

__global__ void bn_apply_relu_kernel(const float* __restrict__ H,
                                     const float* __restrict__ sum,
                                     const float* __restrict__ sumsq,
                                     const float* __restrict__ gamma,
                                     const float* __restrict__ beta,
                                     float* __restrict__ outF,
                                     unsigned short* __restrict__ outB,
                                     long long total, int Ncols, float invM) {
  long long i = (long long)blockIdx.x * blockDim.x + threadIdx.x;
  long long stride = (long long)gridDim.x * blockDim.x;
  for (; i < total; i += stride) {
    int j = (int)(i % Ncols);
    float mu = sum[j] * invM;
    float var = sumsq[j] * invM - mu * mu;
    if (var < 0.f) var = 0.f;
    float y = gamma[j] * (H[i] - mu) * rsqrtf(var + BN_EPS) + beta[j];
    y = fmaxf(y, 0.f);
    outF[i] = y;
    if (outB) outB[i] = f32_to_bf16_rne(y);
  }
}

// ---------------- launch ----------------
extern "C" void kernel_launch(void* const* d_in, const int* in_sizes, int n_in,
                              void* d_out, int out_size, void* d_ws, size_t ws_size,
                              hipStream_t stream) {
  (void)in_sizes; (void)n_in; (void)out_size; (void)ws_size;
  const float*     x    = (const float*)d_in[0];
  const long long* ei   = (const long long*)d_in[1];   // [2][E] int64
  const float*     ew   = (const float*)d_in[2];
  const float*     W1a  = (const float*)d_in[3];
  const float*     W1b  = (const float*)d_in[4];
  const float*     b1   = (const float*)d_in[5];
  const float*     g1   = (const float*)d_in[6];
  const float*     be1  = (const float*)d_in[7];
  const float*     W2a  = (const float*)d_in[8];
  const float*     W2b  = (const float*)d_in[9];
  const float*     b2   = (const float*)d_in[10];
  const float*     g2   = (const float*)d_in[11];
  const float*     be2  = (const float*)d_in[12];
  float* out = (float*)d_out;

  const long long* rows = ei;            // edge_index[0] (src / "row")
  const long long* cols = ei + NEDGES;   // edge_index[1] (dst / "col")

  // ---- workspace carve-up (256B aligned) ----
  char* ws = (char*)d_ws;
  size_t off = 0;
  auto alloc = [&](size_t bytes) -> char* {
    char* p = ws + off;
    off += (bytes + 255) & ~(size_t)255;
    return p;
  };
  float* deg      = (float*)alloc(sizeof(float) * NNODES);
  float* dinv     = (float*)alloc(sizeof(float) * NNODES);
  float* wedge    = (float*)alloc(sizeof(float) * NEDGES);
  unsigned short* x_bf   = (unsigned short*)alloc(2ll * NNODES * D_IN);
  unsigned short* tx1_bf = (unsigned short*)alloc(2ll * NNODES * D_IN);
  float* agg      = (float*)alloc(4ll * NNODES * H1);       // agg1 (D=128) then agg2 (D=512)
  float* h1       = (float*)alloc(4ll * NNODES * H1);       // pre-BN -> normalized in place
  unsigned short* h1n_bf = (unsigned short*)alloc(2ll * NNODES * H1);
  unsigned short* tx2_bf = (unsigned short*)alloc(2ll * NNODES * H1);
  float* h2       = (float*)alloc(4ll * NNODES * H2);
  unsigned short* W1aT = (unsigned short*)alloc(2ll * H1 * D_IN);
  unsigned short* W1bT = (unsigned short*)alloc(2ll * H1 * D_IN);
  unsigned short* W2aT = (unsigned short*)alloc(2ll * H2 * H1);
  unsigned short* W2bT = (unsigned short*)alloc(2ll * H2 * H1);
  float* bnsum    = (float*)alloc(sizeof(float) * H1);
  float* bnsumsq  = (float*)alloc(sizeof(float) * H1);

  const int T = 256;
  dim3 blkT(T);
  auto gsBlocks = [](long long n) { long long b = (n + 255) / 256; if (b > 16384) b = 16384; return (unsigned)b; };

  // ---- edge normalization ----
  zero_f32<<<gsBlocks(NNODES), blkT, 0, stream>>>(deg, NNODES);
  deg_kernel<<<(NEDGES + T - 1) / T, blkT, 0, stream>>>(rows, ew, deg, NEDGES);
  dinv_kernel<<<(NNODES + T - 1) / T, blkT, 0, stream>>>(deg, dinv, NNODES);
  wedge_kernel<<<(NEDGES + T - 1) / T, blkT, 0, stream>>>(rows, cols, ew, dinv, wedge, NEDGES);

  // ---- layer 1 ----
  cvt_bf16_kernel<<<gsBlocks((long long)NNODES * D_IN), blkT, 0, stream>>>(
      x, x_bf, (long long)NNODES * D_IN);
  zero_f32<<<gsBlocks((long long)NNODES * D_IN), blkT, 0, stream>>>(agg, (long long)NNODES * D_IN);
  spmm_edge_atomic<<<(NEDGES * 32 + T - 1) / T, blkT, 0, stream>>>(
      rows, cols, wedge, x, agg, NEDGES, D_IN);
  finalize_tx_kernel<<<gsBlocks((long long)NNODES * D_IN), blkT, 0, stream>>>(
      agg, x, tx1_bf, (long long)NNODES * D_IN, CHEB_LOOP);
  transpose_bf16_kernel<<<gsBlocks((long long)D_IN * H1), blkT, 0, stream>>>(W1a, W1aT, D_IN, H1);
  transpose_bf16_kernel<<<gsBlocks((long long)D_IN * H1), blkT, 0, stream>>>(W1b, W1bT, D_IN, H1);
  {
    long long waves = (long long)(H1 / 64) * ((NNODES + 31) / 32);
    gemm_dualK_wmma<<<(unsigned)((waves * 32 + T - 1) / T), blkT, 0, stream>>>(
        x_bf, W1aT, tx1_bf, W1bT, b1, h1, NNODES, H1, D_IN);
  }
  zero_f32<<<(H1 + T - 1) / T, blkT, 0, stream>>>(bnsum, H1);
  zero_f32<<<(H1 + T - 1) / T, blkT, 0, stream>>>(bnsumsq, H1);
  {
    dim3 grid((H1 + T - 1) / T, (NNODES + 1023) / 1024);
    bn_stats_kernel<<<grid, blkT, 0, stream>>>(h1, bnsum, bnsumsq, NNODES, H1, 1024);
  }
  bn_apply_relu_kernel<<<gsBlocks((long long)NNODES * H1), blkT, 0, stream>>>(
      h1, bnsum, bnsumsq, g1, be1, h1, h1n_bf,
      (long long)NNODES * H1, H1, 1.0f / NNODES);

  // ---- layer 2 ----
  zero_f32<<<gsBlocks((long long)NNODES * H1), blkT, 0, stream>>>(agg, (long long)NNODES * H1);
  spmm_edge_atomic<<<(NEDGES * 32 + T - 1) / T, blkT, 0, stream>>>(
      rows, cols, wedge, h1, agg, NEDGES, H1);
  finalize_tx_kernel<<<gsBlocks((long long)NNODES * H1), blkT, 0, stream>>>(
      agg, h1, tx2_bf, (long long)NNODES * H1, CHEB_LOOP);
  transpose_bf16_kernel<<<gsBlocks((long long)H1 * H2), blkT, 0, stream>>>(W2a, W2aT, H1, H2);
  transpose_bf16_kernel<<<gsBlocks((long long)H1 * H2), blkT, 0, stream>>>(W2b, W2bT, H1, H2);
  {
    long long waves = (long long)((H2 + 63) / 64) * ((NNODES + 31) / 32);
    gemm_dualK_wmma<<<(unsigned)((waves * 32 + T - 1) / T), blkT, 0, stream>>>(
        h1n_bf, W2aT, tx2_bf, W2bT, b2, h2, NNODES, H2, H1);
  }
  zero_f32<<<(H1 + T - 1) / T, blkT, 0, stream>>>(bnsum, H1);
  zero_f32<<<(H1 + T - 1) / T, blkT, 0, stream>>>(bnsumsq, H1);
  {
    dim3 grid((H2 + T - 1) / T, (NNODES + 1023) / 1024);
    bn_stats_kernel<<<grid, blkT, 0, stream>>>(h2, bnsum, bnsumsq, NNODES, H2, 1024);
  }
  bn_apply_relu_kernel<<<gsBlocks((long long)NNODES * H2), blkT, 0, stream>>>(
      h2, bnsum, bnsumsq, g2, be2, out, (unsigned short*)nullptr,
      (long long)NNODES * H2, H2, 1.0f / NNODES);
}
